// DOPAMINEm_44375602102724
// MI455X (gfx1250) — matrix operand order repaired and checked by reading, not compiled, not run on hardware
//
#include <hip/hip_runtime.h>

#define Bn 2
#define Hn 192
#define Wn 192
#define En 10
#define Fn 64
#define NSn 10
#define NMIDn 3
#define BHW (Bn*Hn*Wn)

typedef __attribute__((ext_vector_type(16))) _Float16 v16h;
typedef __attribute__((ext_vector_type(8)))  float    v8f;

__device__ __forceinline__ float clampf(float v, float lo, float hi) {
    return fminf(fmaxf(v, lo), hi);
}

#if __has_builtin(__builtin_amdgcn_tensor_load_to_lds)
#define USE_TDM 1
#else
#define USE_TDM 0
#endif

#if USE_TDM
typedef unsigned int u32x4 __attribute__((ext_vector_type(4)));
typedef int          i32x4 __attribute__((ext_vector_type(4)));
typedef int          i32x8 __attribute__((ext_vector_type(8)));

// 1-D contiguous TDM copy: n8 elements of 8 bytes, global -> LDS byte offset.
// D# per CDNA5 ISA ch.8: group0 {count=1, lds_addr, global_addr, type=2},
// group1 {data_size=3, tensor_dim0=tile_dim0=stride=n8, tensor_dim1=1}.
__device__ __forceinline__ void tdm_load_1d(unsigned lds_addr, const void* gptr,
                                            unsigned n8) {
    unsigned long long ga = (unsigned long long)gptr;
    u32x4 g0;
    g0[0] = 1u;                                            // count=1 (valid)
    g0[1] = lds_addr;                                      // LDS byte address
    g0[2] = (unsigned)(ga & 0xffffffffu);                  // global addr lo
    g0[3] = (unsigned)((ga >> 32) & 0x01ffffffu) | (2u << 30); // addr hi | type=2
    i32x8 g1;
    g1[0] = (int)(3u << 16);                               // data_size=3 (8B)
    g1[1] = (int)((n8 & 0xffffu) << 16);                   // tensor_dim0 lo16
    g1[2] = (int)((n8 >> 16) | (1u << 16));                // td0 hi16 | tensor_dim1=1
    g1[3] = (int)((n8 & 0xffffu) << 16);                   // tile_dim0 = n8
    g1[4] = 0;                                             // tile_dim1/2 unused
    g1[5] = (int)n8;                                       // tensor_dim0_stride lo32
    g1[6] = 0;
    g1[7] = 0;
    i32x4 z4 = {0, 0, 0, 0};
#if __clang_major__ >= 23
    i32x8 z8 = {0, 0, 0, 0, 0, 0, 0, 0};
    __builtin_amdgcn_tensor_load_to_lds(g0, g1, z4, z4, z8, 0);
#else
    __builtin_amdgcn_tensor_load_to_lds(g0, g1, z4, z4, 0);
#endif
}
#endif // USE_TDM

// ---------------------------------------------------------------------------
// Weight prep: f32 (sm, ky, kx, ci, co) -> f16 [z][sm][tap][co][ci]
// ---------------------------------------------------------------------------
__global__ void prep_w_kernel(const float* __restrict__ wm,
                              const float* __restrict__ wp,
                              _Float16* __restrict__ w16, int n) {
    int idx = blockIdx.x * blockDim.x + threadIdx.x;
    if (idx >= n) return;
    int z   = blockIdx.y;
    int sm  = idx / (9 * Fn * Fn);
    int r   = idx % (9 * Fn * Fn);
    int tap = r / (Fn * Fn);
    int rc2 = r % (Fn * Fn);
    int co  = rc2 / Fn;
    int ci  = rc2 % Fn;
    const float* w = z ? wp : wm;
    w16[(size_t)z * n + idx] =
        (_Float16)w[(((size_t)sm * 9 + tap) * Fn + ci) * Fn + co];
}

// ---------------------------------------------------------------------------
// 300-iteration per-pixel exponential fit (pure VALU / transcendental)
// ---------------------------------------------------------------------------
__global__ void iter_kernel(const float* __restrict__ b,
                            const float* __restrict__ tes,
                            float* __restrict__ x) {
    int p = blockIdx.x * blockDim.x + threadIdx.x;
    if (p >= BHW) return;
    int bb = p / (Hn * Wn);
    float bv[En], tv[En];
#pragma unroll
    for (int j = 0; j < En; ++j) {
        bv[j] = b[(size_t)p * En + j];
        tv[j] = tes[bb * En + j];
    }
    float m0 = bv[0];
#pragma unroll
    for (int j = 1; j < En; ++j) m0 = fmaxf(m0, bv[j]);
    m0 = clampf(m0, 0.f, 3.f);
    float p2 = 1.0f;
    for (int it = 0; it < 300; ++it) {
        float gm = 0.f, gp = 0.f;
#pragma unroll
        for (int j = 0; j < En; ++j) {
            float e = __expf(-tv[j] * p2);
            float r = fmaf(m0, e, -bv[j]);
            gm = fmaf(e, r, gm);
            gp = fmaf(-tv[j] * m0 * e, r, gp);
        }
        // 2 * mean = 0.2 * sum
        m0 = clampf(m0 - 0.2f * gm, 0.f, 3.f);
        p2 = clampf(p2 - 0.2f * gp, 0.f, 10.f);
    }
    x[p * 2]     = m0;
    x[p * 2 + 1] = p2;
}

// ---------------------------------------------------------------------------
// conv_in: 1 -> 64 channels, 3x3 SAME, relu, f32 in -> f16 NHWC64 out
// ---------------------------------------------------------------------------
__global__ void conv_in_kernel(const float* __restrict__ x,
                               const float* __restrict__ w_m,
                               const float* __restrict__ w_p,
                               const float* __restrict__ bias_m,
                               const float* __restrict__ bias_p,
                               _Float16* __restrict__ h_out) {
    int z = blockIdx.z;
    const float* w    = z ? w_p : w_m;
    const float* bias = z ? bias_p : bias_m;
    int p = blockIdx.x * blockDim.x + threadIdx.x;
    if (p >= BHW) return;
    int bb  = p / (Hn * Wn);
    int rem = p % (Hn * Wn);
    int y   = rem / Wn;
    int xx  = rem % Wn;
    float acc[Fn];
#pragma unroll
    for (int co = 0; co < Fn; ++co) acc[co] = bias[co];
    for (int ky = 0; ky < 3; ++ky) {
        int gy = y + ky - 1;
        if (gy < 0 || gy >= Hn) continue;
        for (int kx = 0; kx < 3; ++kx) {
            int gx = xx + kx - 1;
            if (gx < 0 || gx >= Wn) continue;
            float xv = x[(((size_t)bb * Hn + gy) * Wn + gx) * 2 + z];
            const float* wt = w + (ky * 3 + kx) * Fn;
#pragma unroll
            for (int co = 0; co < Fn; ++co) acc[co] = fmaf(xv, wt[co], acc[co]);
        }
    }
    _Float16* op = h_out + ((size_t)z * BHW + p) * Fn;
#pragma unroll
    for (int co = 0; co < Fn; ++co) op[co] = (_Float16)fmaxf(acc[co], 0.0f);
}

// ---------------------------------------------------------------------------
// conv_mid: 64 -> 64 channels, 3x3 SAME, relu.  WMMA f16 (f32 accumulate).
// Workgroup = one image row (192 px) = 12 waves x 16 px.
// LDS: weights [9][co=64][ci=64] f16 (72KB, byte offset 0)
//    + rows [3][194][64] f16 (74.5KB, byte offset 73728), staged by TDM.
// Compute: 18 steps (9 taps x 2 k-halves), double-buffered fragment loads so
// LDS latency overlaps the matrix pipe instead of s_wait_dscnt 0 per WMMA.
// ---------------------------------------------------------------------------
__global__ __launch_bounds__(384, 1)
void conv_mid_wmma(const _Float16* __restrict__ h_in,
                   _Float16* __restrict__ h_out,
                   const _Float16* __restrict__ wt_m,
                   const _Float16* __restrict__ wt_p,
                   const float* __restrict__ bias_m,
                   const float* __restrict__ bias_p) {
    extern __shared__ char smem[];
    _Float16* wts  = (_Float16*)smem;             // 36864 halves
    _Float16* rows = (_Float16*)(smem + 73728);   // 37248 halves

    const int z   = blockIdx.z;
    const int by  = blockIdx.y;
    const int bb  = by / Hn;
    const int y   = by % Hn;
    const int tid = threadIdx.x;
    const int wv  = tid >> 5;

    const _Float16* hin  = h_in + (size_t)z * BHW * Fn;
    const _Float16* wt   = z ? wt_p : wt_m;
    const float*    bias = z ? bias_p : bias_m;

#if USE_TDM
    // Wave 0 issues 4 TDM descriptors (weights + up to 3 rows); everyone else
    // zero-fills the halo in parallel with the DMA.
    if (wv == 0) {
        tdm_load_1d(0u, wt, 9216u);   // 73728 B of weights
        for (int r = 0; r < 3; ++r) {
            int gy = y + r - 1;
            if (gy >= 0 && gy < Hn) {
                tdm_load_1d(73728u + (unsigned)((r * 194 + 1) * 128),
                            hin + ((size_t)(bb * Hn + gy) * Wn) * Fn, 3072u);
            }
        }
    }
    // zero halo columns (col 0 and 193 of each LDS row)
    if (tid < 192) {
        int r = tid / 64;
        int c = (tid >> 5) & 1;
        int j = tid & 31;
        ((unsigned*)(rows + (size_t)(r * 194 + (c ? 193 : 0)) * Fn))[j] = 0u;
    }
    // zero out-of-bounds rows (image top/bottom blocks only)
    if (y == 0) {
        unsigned* l32 = (unsigned*)rows;
        for (int idx = tid; idx < 6208; idx += 384) l32[idx] = 0u;
    }
    if (y == Hn - 1) {
        unsigned* l32 = (unsigned*)(rows + (size_t)2 * 194 * Fn);
        for (int idx = tid; idx < 6208; idx += 384) l32[idx] = 0u;
    }
    if (wv == 0) __builtin_amdgcn_s_wait_tensorcnt(0);
    __syncthreads();
#else
    // Fallback: VGPR round-trip staging.
    {
        const uint4* src = (const uint4*)wt;
        uint4*       dst = (uint4*)wts;
#pragma unroll
        for (int i = 0; i < 12; ++i) dst[tid + i * 384] = src[tid + i * 384];
    }
    for (int r = 0; r < 3; ++r) {
        int  gy    = y + r - 1;
        bool rowok = (gy >= 0) && (gy < Hn);
        const unsigned int* g32 =
            (const unsigned int*)(hin + ((size_t)(bb * Hn + (rowok ? gy : 0)) * Wn) * Fn);
        unsigned int* l32 = (unsigned int*)(rows + r * 194 * Fn);
        for (int idx = tid; idx < 6208; idx += 384) {
            int col = idx >> 5;
            int j   = idx & 31;
            unsigned int v = 0u;
            if (rowok && col >= 1 && col <= Wn) v = g32[(col - 1) * 32 + j];
            l32[idx] = v;
        }
    }
    __syncthreads();
#endif

    const int lane = tid & 31;
    const int px0  = wv * 16;
    const int m    = lane & 15;
    const int hi   = lane >> 4;

    v8f acc[4] = {};
    union AFrag { v16h v; uint4 q[2]; };
    AFrag A[2];
    v16h  Bf[2][4];

    // A fragment 16x32 f16: lane(hi=0) K={0..15}, lane(hi=1) K={16..31}
    // B fragment 32x16 f16: lane n holds ci = ks*32 + hi*16 .. +15 of column co
    auto load_step = [&](int st, int bi) {
        const int tap = st >> 1, ks = st & 1;
        const int ky = tap / 3, kx = tap - ky * 3;
        const _Float16* ap =
            rows + ((size_t)(ky * 194) + px0 + m + kx) * Fn + (ks * 32 + hi * 16);
        A[bi].q[0] = *(const uint4*)(ap);
        A[bi].q[1] = *(const uint4*)(ap + 8);
        const _Float16* bbase =
            wts + (size_t)tap * Fn * Fn + (size_t)m * Fn + (ks * 32 + hi * 16);
        Bf[bi][0] = *(const v16h*)(bbase + 0 * 16 * Fn);
        Bf[bi][1] = *(const v16h*)(bbase + 1 * 16 * Fn);
        Bf[bi][2] = *(const v16h*)(bbase + 2 * 16 * Fn);
        Bf[bi][3] = *(const v16h*)(bbase + 3 * 16 * Fn);
    };

    load_step(0, 0);
#pragma unroll
    for (int st = 0; st < 18; ++st) {
        const int cb = st & 1;
        if (st < 17) load_step(st + 1, cb ^ 1);   // prefetch next step
#pragma unroll
        for (int nt = 0; nt < 4; ++nt) {
            acc[nt] = __builtin_amdgcn_wmma_f32_16x16x32_f16(
                false, A[cb].v, false, Bf[cb][nt], (short)0, acc[nt], false, false);
        }
    }
    __syncthreads();

    // bias + relu + f16; transpose through LDS (reuse weight region) for b128 stores
    _Float16* orow = (_Float16*)smem;   // [192][64]
#pragma unroll
    for (int nt = 0; nt < 4; ++nt) {
        const int   co = nt * 16 + m;
        const float bs = bias[co];
#pragma unroll
        for (int r = 0; r < 8; ++r) {
            const int p = px0 + hi * 8 + r;   // C/D layout: lane>=16 holds M=8..15
            orow[p * Fn + co] = (_Float16)fmaxf(acc[nt][r] + bs, 0.0f);
        }
    }
    __syncthreads();
    {
        const uint4* src = (const uint4*)orow;   // 1536 uint4
        uint4* dst = (uint4*)(h_out + (size_t)z * BHW * Fn +
                              (size_t)(bb * Hn + y) * Wn * Fn);
#pragma unroll
        for (int i = 0; i < 4; ++i) dst[tid + i * 384] = src[tid + i * 384];
    }
}

// ---------------------------------------------------------------------------
// conv_out: 64 -> 1 channel, 3x3 SAME (no relu), writes raw conv value
// ---------------------------------------------------------------------------
__global__ void conv_out_kernel(const _Float16* __restrict__ h,
                                const float* __restrict__ w_m,
                                const float* __restrict__ w_p,
                                const float* __restrict__ bias_m,
                                const float* __restrict__ bias_p,
                                float* __restrict__ conv) {
    int z = blockIdx.z;
    const float* w  = z ? w_p : w_m;
    float acc = (z ? bias_p : bias_m)[0];
    int p = blockIdx.x * blockDim.x + threadIdx.x;
    if (p >= BHW) return;
    int bb  = p / (Hn * Wn);
    int rem = p % (Hn * Wn);
    int y   = rem / Wn;
    int xx  = rem % Wn;
    const _Float16* hz = h + (size_t)z * BHW * Fn;
    for (int ky = 0; ky < 3; ++ky) {
        int gy = y + ky - 1;
        if (gy < 0 || gy >= Hn) continue;
        for (int kx = 0; kx < 3; ++kx) {
            int gx = xx + kx - 1;
            if (gx < 0 || gx >= Wn) continue;
            const _Float16* hp = hz + ((size_t)(bb * Hn + gy) * Wn + gx) * Fn;
            const float*    wt = w + (ky * 3 + kx) * Fn;
#pragma unroll
            for (int ci = 0; ci < Fn; ++ci)
                acc = fmaf((float)hp[ci], wt[ci], acc);
        }
    }
    conv[(size_t)z * BHW + p] = acc;
}

// ---------------------------------------------------------------------------
// Stage update: Dx = rc(x - conv); write xm[2i]; gradient step; write xm[2i+1]
// ---------------------------------------------------------------------------
__global__ void stage_update_kernel(const float* __restrict__ b,
                                    const float* __restrict__ tes,
                                    const float* __restrict__ conv,
                                    float* __restrict__ x,
                                    float* __restrict__ out,
                                    const float* __restrict__ mu,
                                    const float* __restrict__ lm,
                                    int stage) {
    int p = blockIdx.x * blockDim.x + threadIdx.x;
    if (p >= BHW) return;
    int bb = p / (Hn * Wn);
    float x0 = x[p * 2], x1 = x[p * 2 + 1];
    float Dx0 = clampf(x0 - conv[p], 0.f, 3.f);
    float Dx1 = clampf(x1 - conv[BHW + p], 0.f, 10.f);
    out[((size_t)(2 * stage) * BHW + p) * 2]     = Dx0;
    out[((size_t)(2 * stage) * BHW + p) * 2 + 1] = Dx1;
    float gm = 0.f, gp = 0.f;
#pragma unroll
    for (int j = 0; j < En; ++j) {
        float t = tes[bb * En + j];
        float e = __expf(-t * x1);
        float r = fmaf(x0, e, -b[(size_t)p * En + j]);
        gm = fmaf(e, r, gm);
        gp = fmaf(-t * x0 * e, r, gp);
    }
    gm *= 0.1f;
    gp *= 0.1f;
    float mu_ = mu[stage], lm_ = lm[stage];
    float n0 = clampf(x0 - mu_ * (gm + lm_ * (x0 - Dx0)), 0.f, 3.f);
    float n1 = clampf(x1 - mu_ * (gp + lm_ * (x1 - Dx1)), 0.f, 10.f);
    out[((size_t)(2 * stage + 1) * BHW + p) * 2]     = n0;
    out[((size_t)(2 * stage + 1) * BHW + p) * 2 + 1] = n1;
    x[p * 2]     = n0;
    x[p * 2 + 1] = n1;
}

// ---------------------------------------------------------------------------
extern "C" void kernel_launch(void* const* d_in, const int* in_sizes, int n_in,
                              void* d_out, int out_size, void* d_ws, size_t ws_size,
                              hipStream_t stream) {
    (void)in_sizes; (void)n_in; (void)out_size; (void)ws_size;
    const float* b      = (const float*)d_in[0];
    const float* tes    = (const float*)d_in[1];
    const float* mu     = (const float*)d_in[2];
    const float* lm     = (const float*)d_in[3];
    const float* Wm_in  = (const float*)d_in[4];
    const float* bm_in  = (const float*)d_in[5];
    const float* Wm_mid = (const float*)d_in[6];
    const float* bm_mid = (const float*)d_in[7];
    const float* Wm_out = (const float*)d_in[8];
    const float* bm_out = (const float*)d_in[9];
    const float* Wp_in  = (const float*)d_in[10];
    const float* bp_in  = (const float*)d_in[11];
    const float* Wp_mid = (const float*)d_in[12];
    const float* bp_mid = (const float*)d_in[13];
    const float* Wp_out = (const float*)d_in[14];
    const float* bp_out = (const float*)d_in[15];
    float* out = (float*)d_out;

    char* ws = (char*)d_ws;
    float*    x_ws  = (float*)(ws);                     // 589,824 B
    float*    convb = (float*)(ws + 589824);            // 589,824 B
    _Float16* hA    = (_Float16*)(ws + 1179648);        // 18,874,368 B
    _Float16* hB    = (_Float16*)(ws + 20054016);       // 18,874,368 B
    _Float16* w16   = (_Float16*)(ws + 38928384);       // 4,423,680 B

    const int WN = NSn * NMIDn * 9 * Fn * Fn;  // 1,105,920 halves per denoiser
    prep_w_kernel<<<dim3((WN + 255) / 256, 2, 1), 256, 0, stream>>>(
        Wm_mid, Wp_mid, w16, WN);

    iter_kernel<<<dim3((BHW + 255) / 256), 256, 0, stream>>>(b, tes, x_ws);

    const size_t SMEM = 73728 + 74496;   // weights + 3 padded rows (f16)
    for (int s = 0; s < NSn; ++s) {
        conv_in_kernel<<<dim3((BHW + 255) / 256, 1, 2), 256, 0, stream>>>(
            x_ws, Wm_in + s * 576, Wp_in + s * 576,
            bm_in + s * 64, bp_in + s * 64, hA);
        _Float16* cur = hA;
        _Float16* nxt = hB;
        for (int mid = 0; mid < NMIDn; ++mid) {
            const _Float16* wtm = w16 + (size_t)((0 * NSn + s) * NMIDn + mid) * (9 * Fn * Fn);
            const _Float16* wtp = w16 + (size_t)((1 * NSn + s) * NMIDn + mid) * (9 * Fn * Fn);
            conv_mid_wmma<<<dim3(1, Bn * Hn, 2), 384, SMEM, stream>>>(
                cur, nxt, wtm, wtp,
                bm_mid + (s * NMIDn + mid) * Fn, bp_mid + (s * NMIDn + mid) * Fn);
            _Float16* t = cur; cur = nxt; nxt = t;
        }
        conv_out_kernel<<<dim3((BHW + 255) / 256, 1, 2), 256, 0, stream>>>(
            cur, Wm_out + s * 576, Wp_out + s * 576, bm_out + s, bp_out + s, convb);
        stage_update_kernel<<<dim3((BHW + 255) / 256), 256, 0, stream>>>(
            b, tes, convb, x_ws, out, mu, lm, s);
    }
    hipMemcpyAsync(out + (size_t)20 * BHW * 2, x_ws,
                   (size_t)BHW * 2 * sizeof(float),
                   hipMemcpyDeviceToDevice, stream);
}
